// GatedAttentionBasedRNN_28217935135073
// MI455X (gfx1250) — matrix operand order repaired and checked by reading, not compiled, not run on hardware
//
#include <hip/hip_runtime.h>

// ---------------------------------------------------------------------------
// R-NET gated attention RNN for MI455X (gfx1250), wave32 + WMMA bf16.
// B=32, C=800, Q=64, H=256, D2=512, D4=1024, 3H=768.
// Strategy: 32 persistent workgroups (1 per WGP, 256 thr = 8 waves), weights
// column-sliced and pinned in LDS (~252KB/WGP) for the whole 800-step scan
// (staged via global_load_async_to_lds_b128 + s_wait_asynccnt), per-step
// activation broadcast through L2, software grid barrier between phases.
// All GEMMs via v_wmma_f32_16x16x32_bf16 with f32 accumulation; next-step
// passage rows prefetched with global_prefetch during the GEMM window.
// ---------------------------------------------------------------------------

#define BB 32
#define CC 800
#define QQ 64
#define HH 256
#define DD2 512
#define DD4 1024
#define G3 768
#define NWG 32
#define BLOCK 256

typedef __attribute__((ext_vector_type(16))) __bf16 bf16x16;
typedef __attribute__((ext_vector_type(8)))  float  f32x8;

// padded LDS row strides (bytes): +16B skews banks by 4 dwords per row
#define RB512  (512 * 2 + 16)
#define RB1024 (1024 * 2 + 16)
#define RB256  (256 * 2 + 16)

#define SWA_OFF   0
#define SWC_OFF   (SWA_OFF + 16 * RB512)
#define SWG_OFF   (SWC_OFF + 16 * RB512)
#define SWIH_OFF  (SWG_OFF + 32 * RB1024)
#define SWHH_OFF  (SWIH_OFF + 48 * RB1024)
#define SGH_OFF   (SWHH_OFF + 48 * RB256)
#define SGI_OFF   (SGH_OFF + 32 * 48 * 4)
#define SPART_OFF (SGI_OFF + 32 * 48 * 4)
#define SH_OFF    (SPART_OFF + 4 * 256 * 4)
#define SP_OFF    (SH_OFF + 32 * 16 * 4)
#define SV_OFF    (SP_OFF + 32 * 64 * 4)
#define SMEM_TOTAL (SV_OFF + 512 * 4)   // ~252.4 KB < 320 KB LDS per WGP

__device__ __forceinline__ f32x8 wmma_bf16(bf16x16 a, bf16x16 b, f32x8 c) {
    return __builtin_amdgcn_wmma_f32_16x16x32_bf16(false, a, false, b,
                                                   (short)0, c, false, false);
}

// A fragment (16x32 bf16, row-major source): lane holds row (lane%16);
// K chunk at k0+16h..+15 with h = lane/16.
__device__ __forceinline__ bf16x16 load_a_frag(const __bf16* rowk0, int h) {
    union { bf16x16 v; uint4 u[2]; } r;
    r.u[0] = *reinterpret_cast<const uint4*>(rowk0 + 16 * h);
    r.u[1] = *reinterpret_cast<const uint4*>(rowk0 + 16 * h + 8);
    return r.v;
}

// B fragment (32x16 bf16) from an LDS weight slice stored row-major [n][K]
// with padded row stride: lane covers k = k0+16h+e of row nLocal.
__device__ __forceinline__ bf16x16 load_b_lds(const char* slice, int rowBytes,
                                              int nLocal, int kOffElems) {
    union { bf16x16 v; uint4 u[2]; } r;
    const uint4* p = reinterpret_cast<const uint4*>(slice + nLocal * rowBytes +
                                                    kOffElems * 2);
    r.u[0] = p[0];
    r.u[1] = p[1];
    return r.v;
}

// B fragment straight from global row-major weights [N][K]
__device__ __forceinline__ bf16x16 load_b_glb(const __bf16* wrowk0, int h) {
    union { bf16x16 v; uint4 u[2]; } r;
    const uint4* p = reinterpret_cast<const uint4*>(wrowk0 + 16 * h);
    r.u[0] = p[0];
    r.u[1] = p[1];
    return r.v;
}

__device__ __forceinline__ float sigmoidf_(float x) {
    return 1.0f / (1.0f + __expf(-x));
}

// ---- CDNA5 async global->LDS copy (16B per active lane), ASYNCcnt-tracked.
// ldsOff = low 32 bits of the generic shared pointer = LDS byte address.
__device__ __forceinline__ void async_copy16(unsigned ldsOff, const void* g) {
    unsigned long long ga = (unsigned long long)g;
    asm volatile("global_load_async_to_lds_b128 %0, %1, off"
                 :
                 : "v"(ldsOff), "v"(ga)
                 : "memory");
}
__device__ __forceinline__ void async_wait0() {
    asm volatile("s_wait_asynccnt 0x0" ::: "memory");
}

// -------------------------- software grid barrier --------------------------
__device__ __forceinline__ void grid_barrier(unsigned* bar) {
    __syncthreads();
    if (threadIdx.x == 0) {
        unsigned* cnt = bar;
        unsigned* gen = bar + 1;
        __threadfence();
        unsigned g = __hip_atomic_load(gen, __ATOMIC_RELAXED, __HIP_MEMORY_SCOPE_AGENT);
        unsigned arrived =
            __hip_atomic_fetch_add(cnt, 1u, __ATOMIC_ACQ_REL, __HIP_MEMORY_SCOPE_AGENT);
        if (arrived == (unsigned)NWG - 1) {
            __hip_atomic_store(cnt, 0u, __ATOMIC_RELAXED, __HIP_MEMORY_SCOPE_AGENT);
            __hip_atomic_fetch_add(gen, 1u, __ATOMIC_RELEASE, __HIP_MEMORY_SCOPE_AGENT);
        } else {
            while (__hip_atomic_load(gen, __ATOMIC_ACQUIRE, __HIP_MEMORY_SCOPE_AGENT) == g)
                __builtin_amdgcn_s_sleep(2);
        }
        __threadfence();
    }
    __syncthreads();
}

// ------------------------------- small kernels ------------------------------
__global__ void k_to_bf16(const float* __restrict__ src, __bf16* __restrict__ dst,
                          long n) {
    long i = (long)blockIdx.x * blockDim.x + threadIdx.x;
    if (i < n) dst[i] = (__bf16)src[i];
}

__global__ void k_init_bar(unsigned* bar) {
    if (threadIdx.x < 2) bar[threadIdx.x] = 0u;
}

// w_q = q_emb @ Wq^T : [2048,512] x [512,512], one 16x16 tile per wave.
__global__ void __launch_bounds__(BLOCK) k_gemm_wq(const __bf16* __restrict__ A,
                                                   const __bf16* __restrict__ W,
                                                   float* __restrict__ Y) {
    const int M = BB * QQ, N = DD2, K = DD2;
    int wave = threadIdx.x >> 5, lane = threadIdx.x & 31;
    int h = lane >> 4, lm = lane & 15;
    long tile = (long)blockIdx.x * 8 + wave;     // 4096 tiles = 512 blocks * 8
    int nTiles = N >> 4;
    int mt = (int)(tile / nTiles), nt = (int)(tile % nTiles);
    if (mt * 16 >= M) return;
    f32x8 acc = {};
    const __bf16* arow = A + (long)(mt * 16 + lm) * K;
    const __bf16* brow = W + (long)(nt * 16 + lm) * K;
    for (int k0 = 0; k0 < K; k0 += 32) {
        bf16x16 a = load_a_frag(arow + k0, h);
        bf16x16 b = load_b_glb(brow + k0, h);
        acc = wmma_bf16(a, b, acc);
    }
    for (int r = 0; r < 8; ++r)
        Y[(long)(mt * 16 + r + 8 * h) * N + nt * 16 + lm] = acc[r];
}

// ------------------------------ persistent kernel ---------------------------
struct RnetParams {
    const float *q_emb, *c_emb, *v;
    const float *b_ih_f, *b_hh_f, *b_ih_b, *b_hh_b;
    const __bf16 *c_bf;
    const __bf16 *Wa_bf, *Wc_bf, *Wg_bf, *wih_f, *whh_f, *wih_b, *whh_b;
    __bf16 *att_bf, *h_bf_f, *h_bf_b, *ctx_bf, *g_bf;
    float *u, *scores, *ctx_f32, *w_q;
    float *out;
    unsigned *bar;
};

__global__ void __launch_bounds__(BLOCK, 1) k_rnet_scan(RnetParams p) {
    const int tid = threadIdx.x;
    const int wg = blockIdx.x;        // 0..31
    const int wave = tid >> 5;
    const int lane = tid & 31;
    const int h = lane >> 4;          // K-half / M-half selector
    const int lm = lane & 15;
    const int dir = wg >> 4;          // 0 = forward GRU, 1 = backward GRU
    const int j0 = (wg & 15) * 16;    // hidden slice
    const int uc0 = wg * 16;          // u / wct column slice
    const int gc0 = wg * 32;          // Wg column slice

    extern __shared__ char smem[];
    char*  sWa   = smem + SWA_OFF;
    char*  sWc   = smem + SWC_OFF;
    char*  sWg   = smem + SWG_OFF;
    char*  sWih  = smem + SWIH_OFF;
    char*  sWhh  = smem + SWHH_OFF;
    float* sGH   = (float*)(smem + SGH_OFF);    // [32][48] hidden-gate GEMM
    float* sGI   = (float*)(smem + SGI_OFF);    // [32][48] input-gate GEMM
    float* sPart = (float*)(smem + SPART_OFF);  // [4][256] K-split partials
    float* sH    = (float*)(smem + SH_OFF);     // [32][16] local hidden state
    float* sP    = (float*)(smem + SP_OFF);     // [32][64] softmax probs
    float* sV    = (float*)(smem + SV_OFF);     // [512]    attention vector v

    const __bf16* wih = dir ? p.wih_b : p.wih_f;
    const __bf16* whh = dir ? p.whh_b : p.whh_f;
    const float*  bih = dir ? p.b_ih_b : p.b_ih_f;
    const float*  bhh = dir ? p.b_hh_b : p.b_hh_f;
    __bf16* hbf = dir ? p.h_bf_b : p.h_bf_f;

    // ---- stage weight column-slices into LDS (resident for all 800 steps)
    // using CDNA5 async global->LDS B128 transfers (no VGPR bounce).
    auto stage = [&](char* dst, int rowBytes, const __bf16* src, long srcStride,
                     int nRows, int kElems) {
        int nChunks = kElems >> 3;  // 16B chunks per row
        for (int r = 0; r < nRows; ++r) {
            const char* s = (const char*)(src + (long)r * srcStride);
            unsigned dOff = (unsigned)(unsigned long long)(dst + r * rowBytes);
            for (int i = tid; i < nChunks; i += BLOCK)
                async_copy16(dOff + i * 16, s + i * 16);
        }
    };
    stage(sWa, RB512, p.Wa_bf + (long)uc0 * DD2, DD2, 16, DD2);
    stage(sWc, RB512, p.Wc_bf + (long)uc0 * DD2, DD2, 16, DD2);
    stage(sWg, RB1024, p.Wg_bf + (long)gc0 * DD4, DD4, 32, DD4);
    for (int gate = 0; gate < 3; ++gate) {
        stage(sWih + gate * 16 * RB1024, RB1024,
              wih + (long)(gate * HH + j0) * DD4, DD4, 16, DD4);
        stage(sWhh + gate * 16 * RB256, RB256,
              whh + (long)(gate * HH + j0) * HH, HH, 16, HH);
    }
    for (int i = tid; i < DD2; i += BLOCK) sV[i] = p.v[i];
    // zero initial state (owned slices)
    for (int i = tid; i < BB * 16; i += BLOCK) {
        int b = i >> 4, jj = i & 15;
        sH[b * 16 + jj] = 0.0f;
        hbf[b * HH + j0 + jj] = (__bf16)0.0f;
        p.att_bf[(long)b * DD2 + dir * HH + j0 + jj] = (__bf16)0.0f;
    }
    async_wait0();          // all async LDS writes landed
    __syncthreads();
    grid_barrier(p.bar);

    // ------------------------------- scan ---------------------------------
    for (int t = 0; t < CC; ++t) {
        // Prefetch next step's passage rows behind the GEMM window.
        if (t + 1 < CC) {
            int b = tid & 31, ch = tid >> 5;   // 8 chunks per row
            __builtin_prefetch(p.c_bf + ((long)b * CC + t + 1) * DD2 + ch * 64, 0, 1);
            __builtin_prefetch(p.c_emb + ((long)b * CC + t + 1) * DD2 + ch * 64, 0, 1);
        }

        // Phase 1: u slice = att@Wa^T + c_emb[t]@Wc^T  (waves 0-1)
        //          gh = h@w_hh^T for owned gates       (waves 2-7) -> LDS
        if (wave < 2) {
            const int mt = wave;
            f32x8 acc = {};
            const __bf16* arow = p.att_bf + (long)(mt * 16 + lm) * DD2;
            for (int k0 = 0; k0 < DD2; k0 += 32) {
                bf16x16 a = load_a_frag(arow + k0, h);
                bf16x16 b = load_b_lds(sWa, RB512, lm, k0 + 16 * h);
                acc = wmma_bf16(a, b, acc);
            }
            const __bf16* crow = p.c_bf + ((long)(mt * 16 + lm) * CC + t) * DD2;
            for (int k0 = 0; k0 < DD2; k0 += 32) {
                bf16x16 a = load_a_frag(crow + k0, h);
                bf16x16 b = load_b_lds(sWc, RB512, lm, k0 + 16 * h);
                acc = wmma_bf16(a, b, acc);
            }
            for (int r = 0; r < 8; ++r) {
                int m = mt * 16 + r + 8 * h;
                p.u[(long)m * DD2 + uc0 + lm] = acc[r];
            }
        } else {
            int tile = wave - 2;             // 0..5
            int mt = tile / 3, nt = tile % 3;
            f32x8 acc = {};
            const __bf16* hrow = hbf + (long)(mt * 16 + lm) * HH;
            for (int k0 = 0; k0 < HH; k0 += 32) {
                bf16x16 a = load_a_frag(hrow + k0, h);
                bf16x16 b = load_b_lds(sWhh, RB256, nt * 16 + lm, k0 + 16 * h);
                acc = wmma_bf16(a, b, acc);
            }
            for (int r = 0; r < 8; ++r) {
                int m = mt * 16 + r + 8 * h;
                sGH[m * 48 + nt * 16 + lm] = acc[r];
            }
        }
        grid_barrier(p.bar);

        // Phase 2: raw attention scores s[b][q] = sum_d v[d]*tanh(wq+u)
        {
            int gw = wg * 8 + wave;                    // 0..255 global waves
            for (int pi = 0; pi < 8; ++pi) {
                int idx = gw * 8 + pi;                 // 0..2047 = (b,q)
                int b = idx >> 6;
                const float* wqr = p.w_q + (long)idx * DD2;
                const float* ur = p.u + (long)b * DD2;
                float s = 0.0f;
                for (int d = lane; d < DD2; d += 32)
                    s += sV[d] * tanhf(wqr[d] + ur[d]);
                for (int off = 16; off; off >>= 1) s += __shfl_xor(s, off);
                if (lane == 0) p.scores[idx] = s;
            }
        }
        grid_barrier(p.bar);

        // Phase 3: softmax (per-b probs in LDS) + ctx column slice
        if (tid < BB) {
            int b = tid;
            const float* sr = p.scores + b * QQ;
            float m = -1e30f;
            for (int q = 0; q < QQ; ++q) m = fmaxf(m, sr[q]);
            float z = 0.0f;
            for (int q = 0; q < QQ; ++q) {
                float e = __expf(sr[q] - m);
                sP[b * QQ + q] = e;
                z += e;
            }
            float zi = 1.0f / z;
            for (int q = 0; q < QQ; ++q) sP[b * QQ + q] *= zi;
        }
        __syncthreads();
        for (int i = tid; i < BB * 16; i += BLOCK) {
            int b = i >> 4, d = uc0 + (i & 15);
            const float* qe = p.q_emb + (long)b * QQ * DD2 + d;
            float acc = 0.0f;
            for (int q = 0; q < QQ; ++q) acc += sP[b * QQ + q] * qe[(long)q * DD2];
            p.ctx_f32[b * DD2 + d] = acc;
            p.ctx_bf[b * DD2 + d] = (__bf16)acc;
        }
        grid_barrier(p.bar);

        // Phase 4: g = sigmoid(sc@Wg^T)*sc for owned 32 columns.
        // 4 tiles x K-split-2 over 8 waves; kh=0 -> passage K, kh=1 -> ctx K.
        {
            int mt = wave >> 2, nt = (wave >> 1) & 1, kh = wave & 1;
            f32x8 acc = {};
            for (int ks = 0; ks < DD2; ks += 32) {
                const __bf16* arow =
                    (kh == 0)
                        ? p.c_bf + ((long)(mt * 16 + lm) * CC + t) * DD2 + ks
                        : p.ctx_bf + (long)(mt * 16 + lm) * DD2 + ks;
                bf16x16 a = load_a_frag(arow, h);
                int k0 = kh * DD2 + ks;
                bf16x16 b = load_b_lds(sWg, RB1024, nt * 16 + lm, k0 + 16 * h);
                acc = wmma_bf16(a, b, acc);
            }
            int ti = mt * 2 + nt;
            if (kh == 1)
                for (int r = 0; r < 8; ++r) sPart[ti * 256 + r * 32 + lane] = acc[r];
            __syncthreads();
            if (kh == 0) {
                for (int r = 0; r < 8; ++r) {
                    float y = acc[r] + sPart[ti * 256 + r * 32 + lane];
                    int m = mt * 16 + r + 8 * h;
                    int col = gc0 + nt * 16 + lm;
                    float scv = (col < DD2)
                                    ? p.c_emb[((long)m * CC + t) * DD2 + col]
                                    : p.ctx_f32[(long)m * DD2 + col - DD2];
                    float gv = scv * sigmoidf_(y);
                    p.g_bf[(long)m * DD4 + col] = (__bf16)gv;
                }
            }
        }
        grid_barrier(p.bar);

        // Phase 5: gi = g@w_ih^T (waves 0-5) + GRU pointwise for 16 hidden.
        if (wave < 6) {
            int mt = wave / 3, nt = wave % 3;
            f32x8 acc = {};
            const __bf16* grow = p.g_bf + (long)(mt * 16 + lm) * DD4;
            for (int k0 = 0; k0 < DD4; k0 += 32) {
                bf16x16 a = load_a_frag(grow + k0, h);
                bf16x16 b = load_b_lds(sWih, RB1024, nt * 16 + lm, k0 + 16 * h);
                acc = wmma_bf16(a, b, acc);
            }
            for (int r = 0; r < 8; ++r) {
                int m = mt * 16 + r + 8 * h;
                sGI[m * 48 + nt * 16 + lm] = acc[r];
            }
        }
        __syncthreads();
        for (int i = tid; i < BB * 16; i += BLOCK) {
            int b = i >> 4, jj = i & 15;
            float gr = sGI[b * 48 + jj] + bih[j0 + jj] +
                       sGH[b * 48 + jj] + bhh[j0 + jj];
            float gz = sGI[b * 48 + 16 + jj] + bih[HH + j0 + jj] +
                       sGH[b * 48 + 16 + jj] + bhh[HH + j0 + jj];
            float r_ = sigmoidf_(gr);
            float z_ = sigmoidf_(gz);
            float hn = sGH[b * 48 + 32 + jj] + bhh[2 * HH + j0 + jj];
            float n_ = tanhf(sGI[b * 48 + 32 + jj] + bih[2 * HH + j0 + jj] + r_ * hn);
            float h2 = (1.0f - z_) * n_ + z_ * sH[b * 16 + jj];
            sH[b * 16 + jj] = h2;
            hbf[b * HH + j0 + jj] = (__bf16)h2;
            p.att_bf[(long)b * DD2 + dir * HH + j0 + jj] = (__bf16)h2;
            p.out[((long)b * CC + t) * DD2 + dir * HH + j0 + jj] = h2;
        }
        grid_barrier(p.bar);
    }
}

// ------------------------------- host launcher ------------------------------
extern "C" void kernel_launch(void* const* d_in, const int* in_sizes, int n_in,
                              void* d_out, int out_size, void* d_ws, size_t ws_size,
                              hipStream_t stream) {
    const float* q_emb  = (const float*)d_in[0];
    const float* c_emb  = (const float*)d_in[1];
    const float* Wq     = (const float*)d_in[2];
    const float* Wc     = (const float*)d_in[3];
    const float* Wa     = (const float*)d_in[4];
    const float* Wg     = (const float*)d_in[5];
    const float* v      = (const float*)d_in[6];
    const float* w_ih_f = (const float*)d_in[7];
    const float* w_hh_f = (const float*)d_in[8];
    const float* b_ih_f = (const float*)d_in[9];
    const float* b_hh_f = (const float*)d_in[10];
    const float* w_ih_b = (const float*)d_in[11];
    const float* w_hh_b = (const float*)d_in[12];
    const float* b_ih_b = (const float*)d_in[13];
    const float* b_hh_b = (const float*)d_in[14];

    char* ws = (char*)d_ws;
    auto alloc = [&](size_t bytes) -> char* {
        char* r = ws;
        ws += (bytes + 255) & ~(size_t)255;
        return r;
    };
    unsigned* bar    = (unsigned*)alloc(256);
    __bf16* att_bf   = (__bf16*)alloc((size_t)BB * DD2 * 2);
    __bf16* h_bf_f   = (__bf16*)alloc((size_t)BB * HH * 2);
    __bf16* h_bf_b   = (__bf16*)alloc((size_t)BB * HH * 2);
    __bf16* ctx_bf   = (__bf16*)alloc((size_t)BB * DD2 * 2);
    __bf16* g_bf     = (__bf16*)alloc((size_t)BB * DD4 * 2);
    float*  u        = (float*)alloc((size_t)BB * DD2 * 4);
    float*  scores   = (float*)alloc((size_t)BB * QQ * 4);
    float*  ctx_f32  = (float*)alloc((size_t)BB * DD2 * 4);
    float*  w_q      = (float*)alloc((size_t)BB * QQ * DD2 * 4);
    __bf16* q_bf     = (__bf16*)alloc((size_t)BB * QQ * DD2 * 2);
    __bf16* c_bf     = (__bf16*)alloc((size_t)BB * CC * DD2 * 2);
    __bf16* Wq_bf    = (__bf16*)alloc((size_t)DD2 * DD2 * 2);
    __bf16* Wa_bf    = (__bf16*)alloc((size_t)DD2 * DD2 * 2);
    __bf16* Wc_bf    = (__bf16*)alloc((size_t)DD2 * DD2 * 2);
    __bf16* Wg_bf    = (__bf16*)alloc((size_t)DD4 * DD4 * 2);
    __bf16* wih_f_bf = (__bf16*)alloc((size_t)G3 * DD4 * 2);
    __bf16* whh_f_bf = (__bf16*)alloc((size_t)G3 * HH * 2);
    __bf16* wih_b_bf = (__bf16*)alloc((size_t)G3 * DD4 * 2);
    __bf16* whh_b_bf = (__bf16*)alloc((size_t)G3 * HH * 2);

    auto cv = [&](const float* s, __bf16* d, long n) {
        k_to_bf16<<<(unsigned)((n + 255) / 256), 256, 0, stream>>>(s, d, n);
    };
    cv(q_emb, q_bf, (long)BB * QQ * DD2);
    cv(c_emb, c_bf, (long)BB * CC * DD2);
    cv(Wq, Wq_bf, (long)DD2 * DD2);
    cv(Wa, Wa_bf, (long)DD2 * DD2);
    cv(Wc, Wc_bf, (long)DD2 * DD2);
    cv(Wg, Wg_bf, (long)DD4 * DD4);
    cv(w_ih_f, wih_f_bf, (long)G3 * DD4);
    cv(w_hh_f, whh_f_bf, (long)G3 * HH);
    cv(w_ih_b, wih_b_bf, (long)G3 * DD4);
    cv(w_hh_b, whh_b_bf, (long)G3 * HH);

    k_init_bar<<<1, 32, 0, stream>>>(bar);
    k_gemm_wq<<<512, BLOCK, 0, stream>>>(q_bf, Wq_bf, w_q);   // 4096 WMMA tiles

    RnetParams p;
    p.q_emb = q_emb; p.c_emb = c_emb; p.v = v;
    p.b_ih_f = b_ih_f; p.b_hh_f = b_hh_f; p.b_ih_b = b_ih_b; p.b_hh_b = b_hh_b;
    p.c_bf = c_bf;
    p.Wa_bf = Wa_bf; p.Wc_bf = Wc_bf; p.Wg_bf = Wg_bf;
    p.wih_f = wih_f_bf; p.whh_f = whh_f_bf; p.wih_b = wih_b_bf; p.whh_b = whh_b_bf;
    p.att_bf = att_bf; p.h_bf_f = h_bf_f; p.h_bf_b = h_bf_b;
    p.ctx_bf = ctx_bf; p.g_bf = g_bf;
    p.u = u; p.scores = scores; p.ctx_f32 = ctx_f32; p.w_q = w_q;
    p.out = (float*)d_out;
    p.bar = bar;

    k_rnet_scan<<<NWG, BLOCK, SMEM_TOTAL, stream>>>(p);
    (void)in_sizes; (void)n_in; (void)out_size; (void)ws_size;
}
